// GraphCastProcessor_61864708931617
// MI455X (gfx1250) — compile-verified, matrix operand order checked by reading, not run on hardware
//
#include <hip/hip_runtime.h>
#include <cstddef>
#include <cstdint>

// MI455X (gfx1250) GraphCast processor step.
// bf16 WMMA 16x16x32 (f32 accum) for all four GEMMs, gather fused into A staging,
// SiLU intermediate kept in LDS as bf16, LayerNorm via shfl_xor + LDS reduction.
// Round 4: weight panels staged with GLOBAL_LOAD_ASYNC_TO_LDS_B128 (ASYNCcnt) and
// double-buffered LDS so global->LDS latency hides behind the WMMA chain.

typedef __attribute__((ext_vector_type(16))) __bf16 v16bf;
typedef __attribute__((ext_vector_type(4)))  __bf16 v4bf;
typedef __attribute__((ext_vector_type(8)))  float  v8f;

#define DD 512
#define MT 32       // rows per workgroup
#define KCH 32      // WMMA K step (bf16)
#define THREADS 256 // 8 waves: 2 (M) x 4 (N)
#define NPW 128     // columns per wave
#define NT 8        // 16-wide WMMA tiles per wave

__device__ __forceinline__ v16bf load_frag_bf16(const __bf16* __restrict__ row,
                                                int kbase, int hi) {
  // 16-bit A/B fragment: lanes 0-15 take K {0..7,16..23}, lanes 16-31 take {8..15,24..31}
  const int r0 = kbase + hi * 8;
  const int r1 = kbase + 16 + hi * 8;
  v16bf f;
#pragma unroll
  for (int i = 0; i < 8; ++i) f[i] = row[r0 + i];
#pragma unroll
  for (int i = 0; i < 8; ++i) f[i + 8] = row[r1 + i];
  return f;
}

template <int K1, bool EDGE>
__global__ __launch_bounds__(THREADS)
void fused_mlp_kernel(const float* __restrict__ A0, const float* __restrict__ NF,
                      const int* __restrict__ SRCI, const int* __restrict__ DSTI,
                      const __bf16* __restrict__ W1T,  // [512][K1], n-major
                      const __bf16* __restrict__ W2T,  // [512][512], n-major
                      const float* __restrict__ B1, const float* __restrict__ B2,
                      const float* __restrict__ G,  const float* __restrict__ BT,
                      float* __restrict__ OUT) {
  __shared__ __attribute__((aligned(16))) __bf16 sB[2][DD][40]; // B ping-pong, 80 KB
  __shared__ __attribute__((aligned(16))) __bf16 sA[2][MT][40]; // A ping-pong,  5 KB
  __shared__ __attribute__((aligned(16))) __bf16 sH[MT][DD];    // SiLU hidden, 32 KB
  __shared__ float red_sum[MT][4];
  __shared__ float red_sq[MT][4];
  __shared__ float s_mean[MT];
  __shared__ float s_rstd[MT];

  const int tid   = threadIdx.x;
  const int wave  = tid >> 5;
  const int waveM = wave >> 2;   // 0..1
  const int waveN = wave & 3;    // 0..3
  const int lane  = tid & 31;
  const int lm    = lane & 15;
  const int hi    = lane >> 4;
  const int row0  = blockIdx.x * MT;
  const int mBase = waveM * 16;

  // A-staging mapping: one float4 per thread per chunk
  const int am = tid >> 3;       // 0..31 row
  const int aj = tid & 7;        // 0..7  k-group of 4

  // Async stage of one K-chunk of a transposed weight panel into sB[b].
  // 8 x global_load_async_to_lds_b128 per wave (per lane), tracked by ASYNCcnt.
  auto issueB = [&](const __bf16* W, int rowStride, int kc, int b) {
#pragma unroll
    for (int i = 0; i < 8; ++i) {
      const int idx = tid + i * THREADS;  // 0..2047
      const int n   = idx >> 2;
      const int j4  = idx & 3;
      const uint32_t lds  = (uint32_t)(uintptr_t)&sB[b][n][j4 * 8];
      const uint32_t goff =
          (uint32_t)(((size_t)n * rowStride + kc + j4 * 8) * sizeof(__bf16));
      asm volatile("global_load_async_to_lds_b128 %0, %1, %2"
                   :: "v"(lds), "v"(goff), "s"((uint64_t)(uintptr_t)W)
                   : "memory");
    }
  };

  // Sync stage of one gathered A chunk (fp32 -> bf16) into sA[b].
  auto stageA = [&](int kc, int b) {
    const int seg = kc >> 9;            // concat segment (uniform per chunk)
    const int rg  = row0 + am;
    const float* srow;
    if (EDGE) {
      srow = (seg == 0) ? (A0 + (size_t)rg * DD)
           : (seg == 1) ? (NF + (size_t)SRCI[rg] * DD)
                        : (NF + (size_t)DSTI[rg] * DD);
    } else {
      srow = (seg == 0) ? (A0 + (size_t)rg * DD) : (NF + (size_t)rg * DD);
    }
    const float4 v = *(const float4*)(srow + (kc & 511) + aj * 4);
    v4bf o;
    o[0] = (__bf16)v.x; o[1] = (__bf16)v.y; o[2] = (__bf16)v.z; o[3] = (__bf16)v.w;
    *(v4bf*)&sA[b][am][aj * 4] = o;
  };

  v8f acc[NT] = {};

  // ---------------- GEMM1: h = silu(cat @ W1 + b1) ----------------
  issueB(W1T, K1, 0, 0);      // prologue: chunk 0 in flight
  stageA(0, 0);
  int buf = 0;
  for (int kc = 0; kc < K1; kc += KCH) {
    const bool hasNext = (kc + KCH) < K1;
    if (hasNext) {
      issueB(W1T, K1, kc + KCH, buf ^ 1);
      stageA(kc + KCH, buf ^ 1);
    }
    // retire this chunk's async loads (leave next chunk's 8 in flight)
    if (hasNext) asm volatile("s_wait_asynccnt 0x8" ::: "memory");
    else         asm volatile("s_wait_asynccnt 0x0" ::: "memory");
    __syncthreads();  // all waves' chunk data visible

    const v16bf af = load_frag_bf16(&sA[buf][mBase + lm][0], 0, hi);
#pragma unroll
    for (int t = 0; t < NT; ++t) {
      const v16bf bf = load_frag_bf16(&sB[buf][waveN * NPW + t * 16 + lm][0], 0, hi);
      acc[t] = __builtin_amdgcn_wmma_f32_16x16x32_bf16(false, af, false, bf,
                                                       (short)0, acc[t], false, false);
    }
    __syncthreads();  // reads done; next iteration may overwrite buf^1
    buf ^= 1;
  }

  // bias + SiLU -> sH (bf16 for second GEMM); fast v_rcp instead of exact fdiv
#pragma unroll
  for (int t = 0; t < NT; ++t) {
    const int col = waveN * NPW + t * 16 + lm;
    const float b = B1[col];
#pragma unroll
    for (int r = 0; r < 8; ++r) {
      const int m = mBase + r + hi * 8;
      const float x = acc[t][r] + b;
      const float sig = __builtin_amdgcn_rcpf(1.0f + __expf(-x));
      sH[m][col] = (__bf16)(x * sig);
    }
  }

  // ---------------- GEMM2: y = h @ W2 + b2 ----------------
  v8f acc2[NT] = {};
  issueB(W2T, DD, 0, 0);      // prologue (safe: GEMM1 loop ended with a barrier)
  buf = 0;
  for (int kc = 0; kc < DD; kc += KCH) {
    const bool hasNext = (kc + KCH) < DD;
    if (hasNext) issueB(W2T, DD, kc + KCH, buf ^ 1);
    if (hasNext) asm volatile("s_wait_asynccnt 0x8" ::: "memory");
    else         asm volatile("s_wait_asynccnt 0x0" ::: "memory");
    __syncthreads();  // chunk data + (first iter) sH writes visible

    const v16bf af = load_frag_bf16(&sH[mBase + lm][0], kc, hi);
#pragma unroll
    for (int t = 0; t < NT; ++t) {
      const v16bf bf = load_frag_bf16(&sB[buf][waveN * NPW + t * 16 + lm][0], 0, hi);
      acc2[t] = __builtin_amdgcn_wmma_f32_16x16x32_bf16(false, af, false, bf,
                                                        (short)0, acc2[t], false, false);
    }
    __syncthreads();
    buf ^= 1;
  }

  // bias + per-row LayerNorm statistics
  float psum[8], psq[8];
#pragma unroll
  for (int r = 0; r < 8; ++r) { psum[r] = 0.f; psq[r] = 0.f; }
#pragma unroll
  for (int t = 0; t < NT; ++t) {
    const int col = waveN * NPW + t * 16 + lm;
    const float b = B2[col];
#pragma unroll
    for (int r = 0; r < 8; ++r) {
      const float v = acc2[t][r] + b;
      acc2[t][r] = v;
      psum[r] += v;
      psq[r]  += v * v;
    }
  }
  // reduce across the 16 lanes of each half-wave (xor masks stay within group)
#pragma unroll
  for (int r = 0; r < 8; ++r) {
#pragma unroll
    for (int msk = 1; msk < 16; msk <<= 1) {
      psum[r] += __shfl_xor(psum[r], msk, 32);
      psq[r]  += __shfl_xor(psq[r],  msk, 32);
    }
  }
  if (lm == 0) {
#pragma unroll
    for (int r = 0; r < 8; ++r) {
      red_sum[mBase + r + hi * 8][waveN] = psum[r];
      red_sq[mBase + r + hi * 8][waveN]  = psq[r];
    }
  }
  __syncthreads();
  if (tid < MT) {
    float s = 0.f, q = 0.f;
#pragma unroll
    for (int w = 0; w < 4; ++w) { s += red_sum[tid][w]; q += red_sq[tid][w]; }
    const float mean = s * (1.0f / DD);
    const float var  = q * (1.0f / DD) - mean * mean;
    s_mean[tid] = mean;
    s_rstd[tid] = rsqrtf(var + 1e-5f);
  }
  __syncthreads();

  // normalize * g + bt + residual, write straight from accumulators
#pragma unroll
  for (int t = 0; t < NT; ++t) {
    const int col = waveN * NPW + t * 16 + lm;
    const float gc = G[col];
    const float bc = BT[col];
#pragma unroll
    for (int r = 0; r < 8; ++r) {
      const int m  = mBase + r + hi * 8;
      const int rg = row0 + m;
      const float res = EDGE ? A0[(size_t)rg * DD + col] : NF[(size_t)rg * DD + col];
      OUT[(size_t)rg * DD + col] =
          (acc2[t][r] - s_mean[m]) * s_rstd[m] * gc + bc + res;
    }
  }
}

// Transpose-convert: src [L][K][512] fp32 -> dst [L][512][K] bf16 (n-major rows)
__global__ void cvt_t_f32_bf16_kernel(const float* __restrict__ s, __bf16* __restrict__ d,
                                      int K, size_t total) {
  size_t i = (size_t)blockIdx.x * blockDim.x + threadIdx.x;
  const size_t stride = (size_t)gridDim.x * blockDim.x;
  const size_t per = (size_t)K * DD;
  for (; i < total; i += stride) {
    const size_t l = i / per;
    const size_t r = i - l * per;
    const int k = (int)(r >> 9);       // r / 512
    const int n = (int)(r & 511);
    d[l * per + (size_t)n * K + k] = (__bf16)s[i];
  }
}

__global__ void zero_f32_kernel(float* __restrict__ p, size_t n) {
  size_t i = (size_t)blockIdx.x * blockDim.x + threadIdx.x;
  const size_t stride = (size_t)gridDim.x * blockDim.x;
  for (; i < n; i += stride) p[i] = 0.0f;
}

__global__ void scatter_add_kernel(const float* __restrict__ ef,
                                   const int* __restrict__ dsti,
                                   float* __restrict__ agg) {
  const int e = blockIdx.x;
  const int d = dsti[e];
  const float* r = ef + (size_t)e * DD;
  float* o = agg + (size_t)d * DD;
  for (int c = threadIdx.x; c < DD; c += blockDim.x) atomicAdd(&o[c], r[c]);
}

extern "C" void kernel_launch(void* const* d_in, const int* in_sizes, int n_in,
                              void* d_out, int out_size, void* d_ws, size_t ws_size,
                              hipStream_t stream) {
  (void)in_sizes; (void)n_in; (void)out_size; (void)ws_size;
  const int L = 4, E = 65536, N = 16384;

  const float* efeat = (const float*)d_in[0];
  const float* nfeat = (const float*)d_in[1];
  const int*   src   = (const int*)d_in[2];
  const int*   dsti  = (const int*)d_in[3];
  const float* ew1   = (const float*)d_in[4];
  const float* eb1   = (const float*)d_in[5];
  const float* ew2   = (const float*)d_in[6];
  const float* eb2   = (const float*)d_in[7];
  const float* eg    = (const float*)d_in[8];
  const float* ebn   = (const float*)d_in[9];
  const float* nw1   = (const float*)d_in[10];
  const float* nb1   = (const float*)d_in[11];
  const float* nw2   = (const float*)d_in[12];
  const float* nb2   = (const float*)d_in[13];
  const float* ng    = (const float*)d_in[14];
  const float* nbn   = (const float*)d_in[15];

  const size_t c_ew1 = (size_t)L * 3 * DD * DD;
  const size_t c_ew2 = (size_t)L * DD * DD;
  const size_t c_nw1 = (size_t)L * 2 * DD * DD;
  const size_t c_nw2 = (size_t)L * DD * DD;

  char* ws = (char*)d_ws;
  __bf16* ew1b = (__bf16*)ws;                    ws += c_ew1 * sizeof(__bf16);
  __bf16* ew2b = (__bf16*)ws;                    ws += c_ew2 * sizeof(__bf16);
  __bf16* nw1b = (__bf16*)ws;                    ws += c_nw1 * sizeof(__bf16);
  __bf16* nw2b = (__bf16*)ws;                    ws += c_nw2 * sizeof(__bf16);
  float*  agg  = (float*)ws;                     // N*DD floats

  // bf16 transposed weight copies (recomputed every call; inputs never mutated)
  cvt_t_f32_bf16_kernel<<<2048, THREADS, 0, stream>>>(ew1, ew1b, 3 * DD, c_ew1);
  cvt_t_f32_bf16_kernel<<<2048, THREADS, 0, stream>>>(ew2, ew2b, DD, c_ew2);
  cvt_t_f32_bf16_kernel<<<2048, THREADS, 0, stream>>>(nw1, nw1b, 2 * DD, c_nw1);
  cvt_t_f32_bf16_kernel<<<2048, THREADS, 0, stream>>>(nw2, nw2b, DD, c_nw2);

  // d_out doubles as the fp32 working state: [efeat | nfeat]
  float* efb = (float*)d_out;
  float* nfb = efb + (size_t)E * DD;
  hipMemcpyAsync(efb, efeat, (size_t)E * DD * sizeof(float),
                 hipMemcpyDeviceToDevice, stream);
  hipMemcpyAsync(nfb, nfeat, (size_t)N * DD * sizeof(float),
                 hipMemcpyDeviceToDevice, stream);

  for (int l = 0; l < L; ++l) {
    fused_mlp_kernel<3 * DD, true><<<E / MT, THREADS, 0, stream>>>(
        efb, nfb, src, dsti,
        ew1b + (size_t)l * 3 * DD * DD, ew2b + (size_t)l * DD * DD,
        eb1 + (size_t)l * DD, eb2 + (size_t)l * DD,
        eg + (size_t)l * DD, ebn + (size_t)l * DD, efb);

    zero_f32_kernel<<<1024, THREADS, 0, stream>>>(agg, (size_t)N * DD);
    scatter_add_kernel<<<E, THREADS, 0, stream>>>(efb, dsti, agg);

    fused_mlp_kernel<2 * DD, false><<<N / MT, THREADS, 0, stream>>>(
        agg, nfb, nullptr, nullptr,
        nw1b + (size_t)l * 2 * DD * DD, nw2b + (size_t)l * DD * DD,
        nb1 + (size_t)l * DD, nb2 + (size_t)l * DD,
        ng + (size_t)l * DD, nbn + (size_t)l * DD, nfb);
  }
}
